// Spec_match_30039001268299
// MI455X (gfx1250) — matrix operand order repaired and checked by reading, not compiled, not run on hardware
//
#include <hip/hip_runtime.h>
#include <hip/hip_bf16.h>
#include <math.h>
#include <stdint.h>

typedef __attribute__((ext_vector_type(16))) _Float16 v16h;
typedef __attribute__((ext_vector_type(8)))  _Float16 v8h;
typedef __attribute__((ext_vector_type(8)))  float    v8f;

#define N_FFT   256
#define HOP     64
#define HS      126
#define FREQ    129
#define FT      9        // 9 freq tiles of 16 -> 144 cols, store only 129
#define NB      8
#define LMIX    65536
#define LAUX    262144
#define TMIX    1025
#define TAUX    4097
#define TAUXP   4176     // padded frame count (>= 4175 needed)
#define NW      26
#define NFLAT   (TMIX * FREQ)      // 132225
#define KMED    ((NFLAT - 1) / 2)  // 66112
#define NBIN    2048
#define TILE_SAMP (15 * HOP + N_FFT)   // 1216 samples per 16-frame tile
#define STFT_THREADS 96                // 3 waves: wave w owns ftiles 3w..3w+2

#define AS1 __attribute__((address_space(1)))
#define AS3 __attribute__((address_space(3)))

#if __has_builtin(__builtin_amdgcn_global_load_async_to_lds_b32)
#define HAVE_ASYNC_LDS 1
#else
#define HAVE_ASYNC_LDS 0
#endif

// ---------------- twiddle tables in WMMA B-operand layout ----------------
// B operand (16-bit, 32x16 KxN), k order chosen here: lane L -> N = L&15, Kgroup = 16*(L>>4);
// half-slot j -> K = Kgroup + j.  Table layout: [k(8)][ftile(9)][lane(32)][j(16)]
__global__ void twiddle_init(_Float16* __restrict__ twc, _Float16* __restrict__ tws) {
    int idx = blockIdx.x * blockDim.x + threadIdx.x;
    if (idx >= 8 * FT * 32 * 16) return;
    int j    = idx & 15;
    int lane = (idx >> 4) & 31;
    int ft   = (idx >> 9) % FT;
    int k    = idx / (16 * 32 * FT);
    int n = 32 * k + 16 * (lane >> 4) + j;     // sample index 0..255
    int f = ft * 16 + (lane & 15);             // freq index 0..143
    int r = (n * f) & 255;                     // exact integer phase reduction
    float theta = (float)(M_PI / 128.0) * (float)r;
    twc[idx] = (_Float16)cosf(theta);          // cos(-x) == cos(x)
    tws[idx] = (_Float16)(-sinf(theta));       // rfft imag uses -sin
}

__device__ __forceinline__ int reflect_idx(int p, int L) {
    if (p < 0)  p = -p;
    if (p >= L) p = 2 * L - 2 - p;
    return p;
}

// ---------------- STFT via WMMA ------------------------------------------
// One 96-thread block (3 waves) per (16-frame tile, batch). The 1216
// samples of the tile are staged in LDS as f16 once; wave w computes freq
// tiles {3w, 3w+1, 3w+2} unconditionally -> no EXEC divergence around WMMA.
__global__ void __launch_bounds__(STFT_THREADS)
stft_wmma(const float* __restrict__ x, int L, int T, int Trows,
          const _Float16* __restrict__ twc, const _Float16* __restrict__ tws,
          float* __restrict__ outRe, float* __restrict__ outIm) {
    __shared__ __align__(16) _Float16 tile[TILE_SAMP];
#if HAVE_ASYNC_LDS
    __shared__ __align__(16) float stage[TILE_SAMP];
#endif
    const int tid   = threadIdx.x;
    const int lane  = tid & 31;
    const int wv    = tid >> 5;          // wave id 0..2
    const int M     = lane & 15;
    const int hsel  = lane >> 4;
    const int b     = blockIdx.y;
    const int ttile = blockIdx.x;
    const float* xb = x + (size_t)b * L;
    const int base  = ttile * (16 * HOP) - (N_FFT / 2);

#if HAVE_ASYNC_LDS
    // async DMA global -> LDS (f32 staging), then convert to f16 tile
    for (int s = tid; s < TILE_SAMP; s += STFT_THREADS) {
        int g = reflect_idx(base + s, L);
        __builtin_amdgcn_global_load_async_to_lds_b32(
            (AS1 int*)(uintptr_t)(xb + g),
            (AS3 int*)(uint32_t)(uintptr_t)&stage[s], 0, 0);
    }
#if __has_builtin(__builtin_amdgcn_s_wait_asynccnt)
    __builtin_amdgcn_s_wait_asynccnt(0);
#else
    asm volatile("s_wait_asynccnt 0x0" ::: "memory");
#endif
    __syncthreads();
    for (int s = tid; s < TILE_SAMP; s += STFT_THREADS)
        tile[s] = (_Float16)stage[s];
#else
    for (int s = tid; s < TILE_SAMP; s += STFT_THREADS) {
        int g = reflect_idx(base + s, L);
        tile[s] = (_Float16)xb[g];
    }
#endif
    __syncthreads();

    v8f zero = {};
    v8f accRe[3], accIm[3];
#pragma unroll
    for (int i = 0; i < 3; ++i) { accRe[i] = zero; accIm[i] = zero; }

#pragma unroll
    for (int k = 0; k < 8; ++k) {
        // A operand (16-bit 16x32 MxK), k order chosen here: lane M = L&15, Kgroup = 16*(L>>4);
        // half-slot j -> K = Kgroup + j, contiguous in LDS (matches B table).
        const int off = M * HOP + 32 * k + 16 * hsel;  // in halves
        v8h lo = *(const v8h*)&tile[off];
        v8h hi = *(const v8h*)&tile[off + 8];
        v16h a = __builtin_shufflevector(lo, hi, 0, 1, 2, 3, 4, 5, 6, 7,
                                                 8, 9, 10, 11, 12, 13, 14, 15);
#pragma unroll
        for (int i = 0; i < 3; ++i) {
            const int ft = 3 * wv + i;                 // always < FT, no guard
            const int boff = ((k * FT + ft) * 32 + lane) << 4;
            const v16h bc = *(const v16h*)(twc + boff);
            const v16h bs = *(const v16h*)(tws + boff);
            accRe[i] = __builtin_amdgcn_wmma_f32_16x16x32_f16(
                false, a, false, bc, (short)0, accRe[i], false, false);
            accIm[i] = __builtin_amdgcn_wmma_f32_16x16x32_f16(
                false, a, false, bs, (short)0, accIm[i], false, false);
        }
    }

    // D (f32 16x16): lane N = L&15; VGPR r holds row M = r + 8*(L>>4)
#pragma unroll
    for (int i = 0; i < 3; ++i) {
        const int ft = 3 * wv + i;
        const int f = ft * 16 + (lane & 15);
        if (f < FREQ) {
#pragma unroll
            for (int r = 0; r < 8; ++r) {
                int tt = ttile * 16 + r + 8 * hsel;
                if (tt < Trows) {
                    size_t o = ((size_t)b * Trows + tt) * FREQ + f;
                    float re = (tt < T) ? accRe[i][r] : 0.0f;  // zero-padded spectra
                    float im = (tt < T) ? accIm[i][r] : 0.0f;
                    outRe[o] = re;
                    outIm[o] = im;
                }
            }
        }
    }
}

// ---------------- exact-rank median via two-pass LDS histogram -----------
__global__ void __launch_bounds__(256)
median_kernel(const float* __restrict__ mixRe, const float* __restrict__ mixIm,
              const float* __restrict__ auxRe, const float* __restrict__ auxIm,
              float* __restrict__ med) {
    __shared__ int hist[NBIN];
    __shared__ int nanCnt;
    __shared__ int sBin, sRank;
    const int b = blockIdx.y;
    const int w = blockIdx.x;
    const int tid = threadIdx.x;

    for (int i = tid; i < NBIN; i += 256) hist[i] = 0;
    if (tid == 0) nanCnt = 0;
    __syncthreads();

    const float* mR = mixRe + (size_t)b * TMIX * FREQ;
    const float* mI = mixIm + (size_t)b * TMIX * FREQ;
    const float* aR = auxRe + ((size_t)b * TAUXP + (size_t)w * HS) * FREQ;
    const float* aI = auxIm + ((size_t)b * TAUXP + (size_t)w * HS) * FREQ;

    // pass 1: coarse histogram of sim in [-1, 1]
    for (int i = tid; i < NFLAT; i += 256) {
        if (i + 8192 < NFLAT) {                 // -> global_prefetch_b8
            __builtin_prefetch(&aR[i + 8192], 0, 1);
            __builtin_prefetch(&aI[i + 8192], 0, 1);
        }
        float mr = mR[i], mi = mI[i], ar = aR[i], ai = aI[i];
        float m2 = mr * mr + mi * mi;
        float w2 = ar * ar + ai * ai;
        if (m2 * w2 == 0.0f) { atomicAdd(&nanCnt, 1); continue; }
        float sim = (mr * ar + mi * ai) * rsqrtf(m2) * rsqrtf(w2);
        int bin = (int)((sim + 1.0f) * (NBIN * 0.5f));
        bin = min(max(bin, 0), NBIN - 1);
        atomicAdd(&hist[bin], 1);
    }
    __syncthreads();
    if (tid == 0) {
        int cum = 0, bin = NBIN - 1, rank = 0;
        for (int i = 0; i < NBIN; ++i) {
            int c = hist[i];
            if (cum + c > KMED) { bin = i; rank = KMED - cum; break; }
            cum += c;
        }
        sBin = bin; sRank = rank;
    }
    __syncthreads();
    const int B0 = sBin;
    const int k1 = sRank;
    for (int i = tid; i < NBIN; i += 256) hist[i] = 0;
    __syncthreads();

    const float binW = 2.0f / NBIN;
    const float lo = -1.0f + B0 * binW;
    // pass 2: refine inside the selected bin
    for (int i = tid; i < NFLAT; i += 256) {
        float mr = mR[i], mi = mI[i], ar = aR[i], ai = aI[i];
        float m2 = mr * mr + mi * mi;
        float w2 = ar * ar + ai * ai;
        if (m2 * w2 == 0.0f) continue;
        float sim = (mr * ar + mi * ai) * rsqrtf(m2) * rsqrtf(w2);
        int bin = (int)((sim + 1.0f) * (NBIN * 0.5f));
        bin = min(max(bin, 0), NBIN - 1);
        if (bin != B0) continue;
        int sub = (int)((sim - lo) * ((float)NBIN / binW));
        sub = min(max(sub, 0), NBIN - 1);
        atomicAdd(&hist[sub], 1);
    }
    __syncthreads();
    if (tid == 0) {
        int cum = 0, sub = NBIN - 1;
        for (int i = 0; i < NBIN; ++i) {
            int c = hist[i];
            if (cum + c > k1) { sub = i; break; }
            cum += c;
        }
        float m = lo + ((float)sub + 0.5f) * (binW / NBIN);
        if (nanCnt > 0) m = -__builtin_inff();
        med[b * NW + w] = m;
    }
}

__global__ void argmax_kernel(const float* __restrict__ med, int* __restrict__ best) {
    int b = threadIdx.x;
    if (b < NB) {
        float bv = -__builtin_inff();
        int bi = 0;
        for (int w = 0; w < NW; ++w) {
            float v = med[b * NW + w];
            if (v > bv) { bv = v; bi = w; }
        }
        best[b] = bi;
    }
}

__global__ void __launch_bounds__(256)
gather_kernel(const float* __restrict__ auxRe, const float* __restrict__ auxIm,
              const int* __restrict__ best, float* __restrict__ out) {
    int b = blockIdx.z;
    int i = blockIdx.x * blockDim.x + threadIdx.x;
    if (i >= NFLAT) return;
    int w = best[b];
    size_t src = ((size_t)b * TAUXP + (size_t)w * HS) * FREQ + i;
    out[((size_t)b * 2 + 0) * NFLAT + i] = auxRe[src];
    out[((size_t)b * 2 + 1) * NFLAT + i] = auxIm[src];
}

extern "C" void kernel_launch(void* const* d_in, const int* in_sizes, int n_in,
                              void* d_out, int out_size, void* d_ws, size_t ws_size,
                              hipStream_t stream) {
    (void)in_sizes; (void)n_in; (void)out_size; (void)ws_size;
    const float* mix = (const float*)d_in[0];
    const float* aux = (const float*)d_in[1];
    float* out = (float*)d_out;
    char* ws = (char*)d_ws;

    // workspace layout (total ~43.1 MB)
    const size_t twElems  = 8 * FT * 32 * 16;                    // 36864 halves
    const size_t auxElems = (size_t)NB * TAUXP * FREQ;           // 4,309,632
    const size_t mixElems = (size_t)NB * TMIX * FREQ;            // 1,057,800
    _Float16* twc   = (_Float16*)(ws);
    _Float16* tws   = (_Float16*)(ws + twElems * 2);
    float*    auxRe = (float*)(ws + twElems * 4);
    float*    auxIm = auxRe + auxElems;
    float*    mixRe = auxIm + auxElems;
    float*    mixIm = mixRe + mixElems;
    float*    med   = mixIm + mixElems;
    int*      best  = (int*)(med + NB * NW);

    twiddle_init<<<(int)((twElems + 255) / 256), 256, 0, stream>>>(twc, tws);

    dim3 gAux(TAUXP / 16, NB);                    // 261 x 8 tiles
    dim3 gMix((TMIX + 15) / 16, NB);              // 65 x 8 tiles
    stft_wmma<<<gAux, STFT_THREADS, 0, stream>>>(aux, LAUX, TAUX, TAUXP, twc, tws, auxRe, auxIm);
    stft_wmma<<<gMix, STFT_THREADS, 0, stream>>>(mix, LMIX, TMIX, TMIX,  twc, tws, mixRe, mixIm);

    median_kernel<<<dim3(NW, NB), 256, 0, stream>>>(mixRe, mixIm, auxRe, auxIm, med);
    argmax_kernel<<<1, 32, 0, stream>>>(med, best);
    gather_kernel<<<dim3((NFLAT + 255) / 256, 1, NB), 256, 0, stream>>>(auxRe, auxIm, best, out);
}
